// Network_27702539059327
// MI455X (gfx1250) — matrix operand order repaired and checked by reading, not compiled, not run on hardware
//
#include <hip/hip_runtime.h>
#include <hip/hip_bf16.h>
#include <math.h>

typedef __attribute__((ext_vector_type(16))) __bf16 v16bf;
typedef __attribute__((ext_vector_type(8)))  __bf16 v8bf;
typedef __attribute__((ext_vector_type(8)))  float  v8f;

#define RAYS    1024
#define SAMP    132
#define P_TOTAL (RAYS * SAMP)   // 135168 = 8448 tiles of 16 = 4224 pairs of 32
#define S_LDS   328             // LDS activation row stride (bf16 elems), bank-friendly

// Packed bf16 weight offsets (elements). Tile layout: [nt][kt][lane][16].
enum : int {
  OFF_W0 = 0,        // 64x256   KT=2  NT=16
  OFF_W1 = 16384,    // 256x256  KT=8  NT=16
  OFF_W2 = 81920,
  OFF_W3 = 147456,
  OFF_W4 = 212992,   // 320x256  KT=10 NT=16 (skip concat, gap at k=63)
  OFF_W5 = 294912,
  OFF_W6 = 360448,
  OFF_W7 = 425984,
  OFF_WA = 491520,   // 256x16   KT=8  NT=1  (alpha head, col0 valid)
  OFF_WF = 495616,   // 256x256  KT=8  NT=16 (feat head)
  OFF_WV = 561152,   // 288x128  KT=9  NT=8  (view layer)
  OFF_WR = 598016,   // 128x16   KT=4  NT=1  (rgb head, cols 0..2 valid)
  W_TOTAL = 600064
};

// Workspace byte offsets
#define WS_Z    ((size_t)0)                    // 1024*132 f32
#define WS_XF   ((size_t)540672)               // P*64 bf16
#define WS_DF   ((size_t)17842176)             // P*32 bf16
#define WS_RAW  ((size_t)26492928)             // P*4  f32 (r,g,b,sigma)
#define WS_WP   ((size_t)28655616)             // W_TOTAL bf16

// ---------------- weight repack: fp32 row-major (K x N) -> bf16 WMMA-B tiles
__global__ void prep_w(const float* __restrict__ src, __bf16* __restrict__ dst,
                       int Ksrc, int Nsrc, int KT, int NT, int g, int w) {
  int idx = blockIdx.x * blockDim.x + threadIdx.x;
  int Kd = KT * 32, Nd = NT * 16;
  if (idx >= Kd * Nd) return;
  int k = idx / Nd, n = idx % Nd;
  int sk = (k < g) ? k : (k - w);
  bool valid = ((k < g) || (k >= g + w)) && (sk < Ksrc) && (n < Nsrc);
  float v = valid ? src[(size_t)sk * Nsrc + n] : 0.f;
  int nt = n >> 4, nl = n & 15, kt = k >> 5, kl = k & 31;
  int lane = nl + (((kl >> 3) & 1) ? 16 : 0);
  int e = (kl & 7) + ((kl >= 16) ? 8 : 0);
  dst[((size_t)(nt * KT + kt) * 32 + lane) * 16 + e] = (__bf16)v;
}

// ---------------- per-ray z sampling (merge + literal sample_pdf + bound insert)
__global__ void sample_kernel(const float* __restrict__ inter, float* __restrict__ zbuf) {
  __shared__ float sm[64 * 196];
  int tid = threadIdx.x;
  int r = blockIdx.x * 64 + tid;
  float* zf = &sm[tid * 196];
  float* za = zf + 64;
  float n0 = inter[r * 4 + 0], f0 = inter[r * 4 + 1];
  float n1 = inter[r * 4 + 2], f1 = inter[r * 4 + 3];
  float s0 = (f0 - n0) / 31.f, s1 = (f1 - n1) / 31.f;
  int i0 = 0, i1 = 0;
  for (int k = 0; k < 64; ++k) {
    float v0 = (i0 < 32) ? (n0 + s0 * i0) : 3.0e38f;
    float v1 = (i1 < 32) ? (n1 + s1 * i1) : 3.0e38f;
    if (v0 <= v1) { zf[k] = v0; ++i0; } else { zf[k] = v1; ++i1; }
  }
  for (int i = 0; i < 128; ++i) {
    float u = (float)i / 127.f;
    int ind = 0;
    for (int j = 0; j < 64; ++j) ind += (u >= (float)j / 63.f) ? 1 : 0;
    int below = ind - 1; if (below < 0) below = 0; if (below > 63) below = 63;
    int above = ind;     if (above < 0) above = 0; if (above > 63) above = 63;
    float c0 = (float)below / 63.f, c1 = (float)above / 63.f;
    float den = (c1 - c0 < 1e-5f) ? 1.f : (c1 - c0);
    float t = (u - c0) / den;
    za[i] = zf[below] + t * (zf[above] - zf[below]);
  }
  float bnd[4] = { n0 - 1e-5f, n1 - 1e-5f, f0 + 1e-5f, f1 + 1e-5f };
  int len = 128;
  for (int b = 0; b < 4; ++b) {
    float bv = bnd[b];
    int pos = len;
    while (pos > 0 && za[pos - 1] > bv) { za[pos] = za[pos - 1]; --pos; }
    za[pos] = bv; ++len;
  }
  for (int i = 0; i < SAMP; ++i) zbuf[(size_t)r * SAMP + i] = za[i];
}

// ---------------- per-point positional encodings -> bf16 feature matrices
__global__ void feat_kernel(const float* __restrict__ rays, const float* __restrict__ zbuf,
                            __bf16* __restrict__ Xf, __bf16* __restrict__ Df) {
  int t = blockIdx.x * blockDim.x + threadIdx.x;
  if (t >= P_TOTAL) return;
  int ray = t / SAMP;
  float o[3], d[3];
  for (int c = 0; c < 3; ++c) { o[c] = rays[ray * 6 + c]; d[c] = rays[ray * 6 + 3 + c]; }
  float nd = sqrtf(d[0]*d[0] + d[1]*d[1] + d[2]*d[2]);
  float z = zbuf[t];
  float p[3];
  for (int c = 0; c < 3; ++c) p[c] = (o[c] + d[c] * z / nd) * 0.1f;  // /DIST
  __bf16* X = Xf + (size_t)t * 64;
  for (int c = 0; c < 3; ++c) X[c] = (__bf16)p[c];
  float fr = 1.f;
  for (int l = 0; l < 10; ++l) {
    for (int c = 0; c < 3; ++c) {
      X[3 + 6*l + c]     = (__bf16)sinf(p[c] * fr);
      X[3 + 6*l + 3 + c] = (__bf16)cosf(p[c] * fr);
    }
    fr *= 2.f;
  }
  X[63] = (__bf16)0.f;
  __bf16* D = Df + (size_t)t * 32;
  for (int c = 0; c < 3; ++c) D[c] = (__bf16)d[c];
  fr = 1.f;
  for (int l = 0; l < 4; ++l) {
    for (int c = 0; c < 3; ++c) {
      D[3 + 6*l + c]     = (__bf16)sinf(d[c] * fr);
      D[3 + 6*l + 3 + c] = (__bf16)cosf(d[c] * fr);
    }
    fr *= 2.f;
  }
  for (int c = 27; c < 32; ++c) D[c] = (__bf16)0.f;
}

// ---------------- WMMA fragment helpers
__device__ __forceinline__ float relu1(float v) {
  // single v_med3_f32 clamp (avoids the canonicalize+max pair fmaxf emits)
  return __builtin_amdgcn_fmed3f(v, 0.f, __builtin_inff());
}

__device__ __forceinline__ v16bf load_a(const __bf16* actIn, int kt, int lane) {
  // 16-bit A 16x32 layout: lane<16 -> row=lane, K {0..7,16..23}; lane>=16 -> row=lane-16, K {8..15,24..31}
  int row = lane & 15;
  int kb = kt * 32 + ((lane & 16) ? 8 : 0);
  const __bf16* p = actIn + row * S_LDS + kb;
  v8bf lo = *(const v8bf*)(p);
  v8bf hi = *(const v8bf*)(p + 16);
  return __builtin_shufflevector(lo, hi, 0,1,2,3,4,5,6,7,8,9,10,11,12,13,14,15);
}

__device__ __forceinline__ v16bf load_b(const __bf16* Wl, int KT, int nt, int kt, int lane) {
  return *(const v16bf*)(Wl + (size_t)((nt * KT + kt) * 32 + lane) * 16);
}

#define WMMA_BF16(a, b, c) \
  __builtin_amdgcn_wmma_f32_16x16x32_bf16(false, (a), false, (b), (short)0, (c), false, false)

// dual M-tile (32 rows) x dual N-tile layer: 4 independent WMMA chains per kt
__device__ __forceinline__ void layer2(const __bf16* in, int KT, const __bf16* Wl,
                                       const float* bias, int Nb,
                                       __bf16* out, int NT, bool relu_on, int lane) {
  int col = lane & 15;
  int rOff = (lane & 16) ? 8 : 0;
  for (int nt = 0; nt < NT; nt += 2) {
    int oc0 = nt * 16 + col, oc1 = oc0 + 16;
    float bv0 = (oc0 < Nb) ? bias[oc0] : 0.f;
    float bv1 = (oc1 < Nb) ? bias[oc1] : 0.f;
    v8f c00 = { bv0, bv0, bv0, bv0, bv0, bv0, bv0, bv0 };
    v8f c01 = { bv1, bv1, bv1, bv1, bv1, bv1, bv1, bv1 };
    v8f c10 = c00, c11 = c01;
    for (int kt = 0; kt < KT; ++kt) {
      v16bf a0 = load_a(in, kt, lane);
      v16bf a1 = load_a(in + 16 * S_LDS, kt, lane);
      v16bf b0 = load_b(Wl, KT, nt, kt, lane);
      v16bf b1 = load_b(Wl, KT, nt + 1, kt, lane);
      c00 = WMMA_BF16(a0, b0, c00);
      c10 = WMMA_BF16(a1, b0, c10);
      c01 = WMMA_BF16(a0, b1, c01);
      c11 = WMMA_BF16(a1, b1, c11);
    }
#pragma unroll
    for (int r = 0; r < 8; ++r) {
      float v00 = c00[r], v01 = c01[r], v10 = c10[r], v11 = c11[r];
      if (relu_on) {
        v00 = relu1(v00); v01 = relu1(v01);
        v10 = relu1(v10); v11 = relu1(v11);
      }
      out[(r + rOff) * S_LDS + oc0]        = (__bf16)v00;
      out[(r + rOff) * S_LDS + oc1]        = (__bf16)v01;
      out[(16 + r + rOff) * S_LDS + oc0]   = (__bf16)v10;
      out[(16 + r + rOff) * S_LDS + oc1]   = (__bf16)v11;
    }
  }
}

// dual M-tile, single N-tile head (shared B fragment)
__device__ __forceinline__ void head2(const __bf16* in, int KT, const __bf16* Wl,
                                      float bv, int lane, v8f* o0, v8f* o1) {
  v8f c0 = { bv, bv, bv, bv, bv, bv, bv, bv };
  v8f c1 = c0;
  for (int kt = 0; kt < KT; ++kt) {
    v16bf a0 = load_a(in, kt, lane);
    v16bf a1 = load_a(in + 16 * S_LDS, kt, lane);
    v16bf b = load_b(Wl, KT, 0, kt, lane);
    c0 = WMMA_BF16(a0, b, c0);
    c1 = WMMA_BF16(a1, b, c1);
  }
  *o0 = c0; *o1 = c1;
}

// ---------------- fused NeRF MLP: 2 waves/block, 32 points (2 M-tiles) per wave
__global__ __launch_bounds__(64) void
mlp_kernel(const __bf16* __restrict__ Xf, const __bf16* __restrict__ Df,
           const __bf16* __restrict__ Wp, float* __restrict__ raw,
           const float* b0p, const float* b1p, const float* b2p, const float* b3p,
           const float* b4p, const float* b5p, const float* b6p, const float* b7p,
           const float* abp, const float* fbp, const float* vbp, const float* rbp) {
  __shared__ __bf16 smem[2 * 2 * 32 * S_LDS];
  int wid = threadIdx.x >> 5, lane = threadIdx.x & 31;
  __bf16* bufA = &smem[wid * (2 * 32 * S_LDS)];
  __bf16* bufB = bufA + 32 * S_LDS;
  size_t pbase = ((size_t)blockIdx.x * 2 + wid) * 32;

  // Async fill X features (32 rows x 128B) into bufA[0..63] -- CDNA5 async LDS path.
  // Issued FIRST so they complete first (async loads complete in order).
  for (int it = 0; it < 8; ++it) {
    int c = it * 32 + lane;
    int row = c >> 3, cc = c & 7;                 // 8 x 16B chunks per 64-col row
    unsigned int ldst = (unsigned int)(uintptr_t)bufA +
                        (unsigned int)((row * S_LDS + cc * 8) * 2);
    unsigned long long ga = (unsigned long long)(uintptr_t)(Xf + (pbase + row) * 64 + cc * 8);
    asm volatile("global_load_async_to_lds_b128 %0, %1, off"
                 :: "v"(ldst), "v"(ga) : "memory");
  }
  // Async fill dir features (32 rows x 64B) into bufB cols 256..287 (never clobbered by
  // layer writes, not consumed until the view layer -> overlap with all dense layers).
  for (int it = 0; it < 4; ++it) {
    int c = it * 32 + lane;
    int row = c >> 2, cc = c & 3;                 // 4 x 16B chunks per 32-col row
    unsigned int ldst = (unsigned int)(uintptr_t)bufB +
                        (unsigned int)((row * S_LDS + 256 + cc * 8) * 2);
    unsigned long long ga = (unsigned long long)(uintptr_t)(Df + (pbase + row) * 32 + cc * 8);
    asm volatile("global_load_async_to_lds_b128 %0, %1, off"
                 :: "v"(ldst), "v"(ga) : "memory");
  }
  // Wait only for the 8 X-chunks (the 4 D-chunks may still be in flight).
  asm volatile("s_wait_asynccnt 4" ::: "memory");

  layer2(bufA,      2,  Wp + OFF_W0, b0p, 256, bufB,      16, true, lane);  // L0
  layer2(bufB,      8,  Wp + OFF_W1, b1p, 256, bufA + 64, 16, true, lane);  // L1
  layer2(bufA + 64, 8,  Wp + OFF_W2, b2p, 256, bufB,      16, true, lane);  // L2
  layer2(bufB,      8,  Wp + OFF_W3, b3p, 256, bufA + 64, 16, true, lane);  // L3
  // skip: bufA = [x(64) | h(256)] = 320 cols
  layer2(bufA,      10, Wp + OFF_W4, b4p, 256, bufB,      16, true, lane);  // L4
  layer2(bufB,      8,  Wp + OFF_W5, b5p, 256, bufA + 64, 16, true, lane);  // L5
  layer2(bufA + 64, 8,  Wp + OFF_W6, b6p, 256, bufB,      16, true, lane);  // L6
  layer2(bufB,      8,  Wp + OFF_W7, b7p, 256, bufA + 64, 16, true, lane);  // L7

  // sigma head (256 -> 16, col0 valid), no relu
  {
    float bv = ((lane & 15) == 0) ? abp[0] : 0.f;
    v8f c0, c1;
    head2(bufA + 64, 8, Wp + OFF_WA, bv, lane, &c0, &c1);
    if ((lane & 15) == 0) {
      int rOff = (lane & 16) ? 8 : 0;
#pragma unroll
      for (int r = 0; r < 8; ++r) {
        raw[(pbase + r + rOff) * 4 + 3]      = c0[r];
        raw[(pbase + 16 + r + rOff) * 4 + 3] = c1[r];
      }
    }
  }

  // feat head (256 -> 256), NO relu
  layer2(bufA + 64, 8, Wp + OFF_WF, fbp, 256, bufB, 16, false, lane);

  // make sure dir features have landed before the view layer consumes them
  asm volatile("s_wait_asynccnt 0" ::: "memory");

  // view layer (288 -> 128): reads feat(0..255) + dirs(256..287), relu
  layer2(bufB, 9, Wp + OFF_WV, vbp, 128, bufA, 8, true, lane);

  // rgb head (128 -> 16, cols 0..2 valid), no relu
  {
    int col = lane & 15;
    float bv = (col < 3) ? rbp[col] : 0.f;
    v8f c0, c1;
    head2(bufA, 4, Wp + OFF_WR, bv, lane, &c0, &c1);
    if (col < 3) {
      int rOff = (lane & 16) ? 8 : 0;
#pragma unroll
      for (int r = 0; r < 8; ++r) {
        raw[(pbase + r + rOff) * 4 + col]      = c0[r];
        raw[(pbase + 16 + r + rOff) * 4 + col] = c1[r];
      }
    }
  }
}

// ---------------- alpha compositing
__global__ void composite_kernel(const float* __restrict__ rays, const float* __restrict__ zbuf,
                                 const float* __restrict__ raw, float* __restrict__ out) {
  int r = blockIdx.x * blockDim.x + threadIdx.x;
  if (r >= RAYS) return;
  float dx = rays[r * 6 + 3], dy = rays[r * 6 + 4], dz = rays[r * 6 + 5];
  float nd = sqrtf(dx * dx + dy * dy + dz * dz);
  float T = 1.f, rm = 0.f, gm = 0.f, bm = 0.f, depth = 0.f, acc = 0.f;
  const float* zr = zbuf + (size_t)r * SAMP;
  const float4* rw = (const float4*)(raw) + (size_t)r * SAMP;
  for (int i = 0; i < SAMP; ++i) {
    float zi = zr[i];
    float dist = (i < SAMP - 1) ? (zr[i + 1] - zi) : (1e10f * nd);
    float4 v = rw[i];
    float alpha = 1.f - expf(-fmaxf(v.w, 0.f) * dist);
    float w = alpha * T;
    rm += w / (1.f + expf(-v.x));
    gm += w / (1.f + expf(-v.y));
    bm += w / (1.f + expf(-v.z));
    depth += w * (zi / nd);
    acc += w;
    T *= (1.f - alpha + 1e-10f);
  }
  out[r * 3 + 0] = rm;
  out[r * 3 + 1] = gm;
  out[r * 3 + 2] = bm;
  out[3072 + r] = depth;
  out[4096 + r] = acc;
}

extern "C" void kernel_launch(void* const* d_in, const int* in_sizes, int n_in,
                              void* d_out, int out_size, void* d_ws, size_t ws_size,
                              hipStream_t stream) {
  (void)in_sizes; (void)n_in; (void)out_size; (void)ws_size;
  const float* rays  = (const float*)d_in[0];
  const float* inter = (const float*)d_in[1];

  char* ws = (char*)d_ws;
  float*  zbuf = (float*)(ws + WS_Z);
  __bf16* Xf   = (__bf16*)(ws + WS_XF);
  __bf16* Df   = (__bf16*)(ws + WS_DF);
  float*  rawb = (float*)(ws + WS_RAW);
  __bf16* Wp   = (__bf16*)(ws + WS_WP);

  auto prep = [&](int src_idx, int off, int Ksrc, int Nsrc, int KT, int NT, int g, int w) {
    int n = KT * 32 * NT * 16;
    prep_w<<<(n + 255) / 256, 256, 0, stream>>>((const float*)d_in[src_idx], Wp + off,
                                                Ksrc, Nsrc, KT, NT, g, w);
  };
  prep(2,  OFF_W0, 63,  256, 2,  16, 63,  1);
  prep(4,  OFF_W1, 256, 256, 8,  16, 256, 0);
  prep(6,  OFF_W2, 256, 256, 8,  16, 256, 0);
  prep(8,  OFF_W3, 256, 256, 8,  16, 256, 0);
  prep(10, OFF_W4, 319, 256, 10, 16, 63,  1);
  prep(12, OFF_W5, 256, 256, 8,  16, 256, 0);
  prep(14, OFF_W6, 256, 256, 8,  16, 256, 0);
  prep(16, OFF_W7, 256, 256, 8,  16, 256, 0);
  prep(18, OFF_WA, 256, 1,   8,  1,  256, 0);
  prep(20, OFF_WF, 256, 256, 8,  16, 256, 0);
  prep(22, OFF_WV, 283, 128, 9,  8,  288, 0);
  prep(24, OFF_WR, 128, 3,   4,  1,  128, 0);

  sample_kernel<<<16, 64, 0, stream>>>(inter, zbuf);
  feat_kernel<<<(P_TOTAL + 255) / 256, 256, 0, stream>>>(rays, zbuf, Xf, Df);

  // 8448 M-tiles, 2 per wave, 2 waves per block -> 2112 blocks of 64 threads
  mlp_kernel<<<P_TOTAL / (32 * 2), 64, 0, stream>>>(
      Xf, Df, Wp, rawb,
      (const float*)d_in[3],  (const float*)d_in[5],  (const float*)d_in[7],  (const float*)d_in[9],
      (const float*)d_in[11], (const float*)d_in[13], (const float*)d_in[15], (const float*)d_in[17],
      (const float*)d_in[19], (const float*)d_in[21], (const float*)d_in[23], (const float*)d_in[25]);

  composite_kernel<<<(RAYS + 255) / 256, 256, 0, stream>>>(rays, zbuf, rawb, (float*)d_out);
}
